// BatchTopKSAE_19774029431638
// MI455X (gfx1250) — compile-verified, matrix operand order checked by reading, not run on hardware
//
#include <hip/hip_runtime.h>
#include <hip/hip_bf16.h>
#include <stdint.h>

#ifndef __has_builtin
#define __has_builtin(x) 0
#endif

#define INPUT_DIM 1024
#define DICT_SIZE 16384
#define BATCH 4096
#define TOPK_TOTAL 131072u   // K(32) * BATCH(4096)

typedef __bf16 bf16_t;
typedef __attribute__((ext_vector_type(16))) __bf16 v16bf;
typedef __attribute__((ext_vector_type(8)))  float  v8f;
typedef unsigned int u32x4 __attribute__((ext_vector_type(4)));
typedef int          i32x4 __attribute__((ext_vector_type(4)));
typedef int          i32x8 __attribute__((ext_vector_type(8)));

#if __has_builtin(__builtin_amdgcn_tensor_load_to_lds) && \
    __has_builtin(__builtin_amdgcn_s_wait_tensorcnt)
#define SAE_HAVE_TDM 1
#else
#define SAE_HAVE_TDM 0
#endif

// ---------------------------------------------------------------------------
// helpers
// ---------------------------------------------------------------------------
__device__ __forceinline__ unsigned pack_bf16_2(float a, float b) {
    union { bf16_t h[2]; unsigned u; } t;
    t.h[0] = (bf16_t)a;
    t.h[1] = (bf16_t)b;
    return t.u;
}

// generic shared pointer -> 32-bit LDS byte offset (low 32 bits of flat addr)
__device__ __forceinline__ unsigned lds_off32(const void* p) {
    return (unsigned)(uintptr_t)p;
}

// Load one 16x32 bf16 fragment from an LDS tile, row stride 40 halves (80 B:
// 64 B data + 16 B pad).  Lane l: half = l>>4.
// e0..7  <- K = 8*half..8*half+7   at bytes row*80 + half*16
// e8..15 <- K = 16+8*half..23+8*half at bytes row*80 + 32 + half*16
__device__ __forceinline__ v16bf lds_frag(const bf16_t* base, int row, int half) {
    union U { struct { uint4 a, b; } q; v16bf v; } u;
    const bf16_t* p = base + row * 40 + half * 8;
    u.q.a = *(const uint4*)(p);
    u.q.b = *(const uint4*)(p + 16);
    return u.v;
}

template <int N>
__device__ __forceinline__ void tdm_wait() {
#if SAE_HAVE_TDM
    __builtin_amdgcn_s_wait_tensorcnt((short)N);
#endif
}

// Issue one TDM descriptor: DMA a 128-row x 32-elem bf16 tile from a row-major
// [rows, K] bf16 tensor (tile origin = element offset eoff) into LDS at
// lds_addr, inserting 4 DWORDs of LDS padding per 16 DWORDs (row stride 80 B).
__device__ __forceinline__ void tdm_load_tile(unsigned lds_addr,
                                              const bf16_t* gbase,
                                              unsigned long long eoff,
                                              int K, int rows) {
#if SAE_HAVE_TDM
    unsigned long long ga = (unsigned long long)(uintptr_t)gbase + eoff * 2ull;
    u32x4 g0;
    g0[0] = 1u;                                          // count=1 (valid), no gather
    g0[1] = lds_addr;                                    // lds_addr[63:32 of D#]
    g0[2] = (unsigned)(ga & 0xFFFFFFFFull);              // global_addr lo
    g0[3] = (unsigned)((ga >> 32) & 0x01FFFFFFull) | (2u << 30);  // addr hi | type=2
    i32x8 g1;
    // data_size=1 (2B) | pad_enable | pad_interval=3 (16 dw) | pad_amount=3 (4 dw)
    g1[0] = (int)((1u << 16) | (1u << 20) | (3u << 22) | (3u << 25));
    g1[1] = (int)(((unsigned)K & 0xFFFFu) << 16);                        // tensor_dim0 lo
    g1[2] = (int)(((unsigned)K >> 16) | (((unsigned)rows & 0xFFFFu) << 16)); // dim0 hi | dim1 lo
    g1[3] = (int)(((unsigned)rows >> 16) | (32u << 16));                 // dim1 hi | tile_dim0=32
    g1[4] = (int)(128u);                                                 // tile_dim1=128, tile_dim2=0
    g1[5] = (int)K;                                                      // tensor_dim0_stride lo
    g1[6] = 0;                                                           // stride0 hi | stride1 lo
    g1[7] = 0;                                                           // stride1 hi
    i32x4 gz = {0, 0, 0, 0};
#if defined(__clang_major__) && (__clang_major__ >= 23)
    i32x8 gz8 = {0, 0, 0, 0, 0, 0, 0, 0};
    __builtin_amdgcn_tensor_load_to_lds(g0, g1, gz, gz, gz8, 0);
#else
    __builtin_amdgcn_tensor_load_to_lds(g0, g1, gz, gz, 0);
#endif
#endif
}

// ---------------------------------------------------------------------------
// TDM-pipelined GEMM:  C[M,N] = act( A[M,K] * B[N,K]^T + bias[N] )
// B is pre-converted bf16, staged by the Tensor Data Mover (double-buffered).
// A_PRE: A pre-converted bf16, also staged by TDM.  !A_PRE: A fp32, staged by
// VALU convert (decoder: A = features, produced mid-launch).
// Block tile 128x128, BK=32, 256 threads (8 waves), wave tile 32x64.
// ---------------------------------------------------------------------------
template <bool ENC, bool A_PRE>
__global__ __launch_bounds__(256) void sae_gemm_wmma_tdm(
    const bf16_t* __restrict__ Abf, const float* __restrict__ Afp,
    const bf16_t* __restrict__ Bbf, const float* __restrict__ bias,
    float* __restrict__ C, int M, int N, int K)
{
    const int tid   = threadIdx.x;
    const int lane  = tid & 31;
    const int wave  = tid >> 5;
    const int waveM = wave >> 1;
    const int waveN = wave & 1;
    const int half  = lane >> 4;
    const int l16   = lane & 15;

    const int blockM = blockIdx.y * 128;
    const int blockN = blockIdx.x * 128;

    __shared__ __align__(16) bf16_t As[2][128 * 40];
    __shared__ __align__(16) bf16_t Bs[2][128 * 40];

    v8f acc[2][4] = {};

    const int ldr = tid >> 1;
    const int ldc = (tid & 1) * 16;
    const int nk  = K / 32;
    constexpr int NT = A_PRE ? 2 : 1;   // TDM descriptors per stage

    // prologue: stage 0 into buffer 0
    if (tid == 0) {
        if (A_PRE) tdm_load_tile(lds_off32(&As[0][0]), Abf,
                                 (unsigned long long)blockM * K, K, M);
        tdm_load_tile(lds_off32(&Bs[0][0]), Bbf,
                      (unsigned long long)blockN * K, K, N);
    }

    int cur = 0;
    for (int k = 0; k < nk; ++k) {
        const int kk = k * 32;
        if (tid == 0) {
            if (k + 1 < nk) {   // prefetch next stage into the other buffer
                if (A_PRE) tdm_load_tile(lds_off32(&As[cur ^ 1][0]), Abf,
                                         (unsigned long long)blockM * K + kk + 32, K, M);
                tdm_load_tile(lds_off32(&Bs[cur ^ 1][0]), Bbf,
                              (unsigned long long)blockN * K + kk + 32, K, N);
                tdm_wait<NT>();     // current stage complete; next stage in flight
            } else {
                tdm_wait<0>();
            }
        }
        if (!A_PRE) {   // VALU-stage fp32 A tile (buffer free since last barrier)
            const float4* gp = (const float4*)(Afp + (size_t)(blockM + ldr) * K + kk + ldc);
            float4 f0 = gp[0], f1 = gp[1], f2 = gp[2], f3 = gp[3];
            uint4 q0, q1;
            q0.x = pack_bf16_2(f0.x, f0.y); q0.y = pack_bf16_2(f0.z, f0.w);
            q0.z = pack_bf16_2(f1.x, f1.y); q0.w = pack_bf16_2(f1.z, f1.w);
            q1.x = pack_bf16_2(f2.x, f2.y); q1.y = pack_bf16_2(f2.z, f2.w);
            q1.z = pack_bf16_2(f3.x, f3.y); q1.w = pack_bf16_2(f3.z, f3.w);
            *(uint4*)&As[cur][ldr * 40 + ldc]     = q0;
            *(uint4*)&As[cur][ldr * 40 + ldc + 8] = q1;
        }
        __syncthreads();   // publish staged buffers to all waves

        v16bf af[2], bfg[4];
        #pragma unroll
        for (int fm = 0; fm < 2; ++fm)
            af[fm] = lds_frag(As[cur], waveM * 32 + fm * 16 + l16, half);
        #pragma unroll
        for (int fn = 0; fn < 4; ++fn)
            bfg[fn] = lds_frag(Bs[cur], waveN * 64 + fn * 16 + l16, half);

        #pragma unroll
        for (int fm = 0; fm < 2; ++fm)
            #pragma unroll
            for (int fn = 0; fn < 4; ++fn)
                acc[fm][fn] = __builtin_amdgcn_wmma_f32_16x16x32_bf16(
                    false, af[fm], false, bfg[fn],
                    (short)0, acc[fm][fn], false, false);

        __syncthreads();   // compute done before this buffer is overwritten
        cur ^= 1;
    }

    #pragma unroll
    for (int fn = 0; fn < 4; ++fn) {
        const int col = blockN + waveN * 64 + fn * 16 + l16;
        const float bv = bias[col];
        #pragma unroll
        for (int fm = 0; fm < 2; ++fm) {
            const int rowb = blockM + waveM * 32 + fm * 16 + half * 8;
            #pragma unroll
            for (int r = 0; r < 8; ++r) {
                float v = acc[fm][fn][r] + bv;
                if (ENC) v = (v > 0.0f) ? v : 0.0f;   // forces +0, no -0 bits
                C[(size_t)(rowb + r) * N + col] = v;
            }
        }
    }
}

// ---------------------------------------------------------------------------
// Synchronous fallback GEMM (all-fp32 operands, on-the-fly bf16 convert).
// Used when TDM builtins or workspace for bf16 copies are unavailable.
// ---------------------------------------------------------------------------
template <bool ENC>
__global__ __launch_bounds__(256) void sae_gemm_wmma_sync(
    const float* __restrict__ A, const float* __restrict__ B,
    const float* __restrict__ SUBV, const float* __restrict__ bias,
    float* __restrict__ C, int M, int N, int K)
{
    const int tid   = threadIdx.x;
    const int lane  = tid & 31;
    const int wave  = tid >> 5;
    const int waveM = wave >> 1;
    const int waveN = wave & 1;
    const int half  = lane >> 4;
    const int l16   = lane & 15;

    const int blockM = blockIdx.y * 128;
    const int blockN = blockIdx.x * 128;

    __shared__ __align__(16) bf16_t As[128 * 40];
    __shared__ __align__(16) bf16_t Bs[128 * 40];

    v8f acc[2][4] = {};
    const int ldr = tid >> 1;
    const int ldc = (tid & 1) * 16;

    for (int kk = 0; kk < K; kk += 32) {
        __syncthreads();
        {
            const float4* gp = (const float4*)(A + (size_t)(blockM + ldr) * K + kk + ldc);
            float4 f0 = gp[0], f1 = gp[1], f2 = gp[2], f3 = gp[3];
            if (ENC) {
                const float4* sp = (const float4*)(SUBV + kk + ldc);
                float4 s0 = sp[0], s1 = sp[1], s2 = sp[2], s3 = sp[3];
                f0.x -= s0.x; f0.y -= s0.y; f0.z -= s0.z; f0.w -= s0.w;
                f1.x -= s1.x; f1.y -= s1.y; f1.z -= s1.z; f1.w -= s1.w;
                f2.x -= s2.x; f2.y -= s2.y; f2.z -= s2.z; f2.w -= s2.w;
                f3.x -= s3.x; f3.y -= s3.y; f3.z -= s3.z; f3.w -= s3.w;
            }
            uint4 q0, q1;
            q0.x = pack_bf16_2(f0.x, f0.y); q0.y = pack_bf16_2(f0.z, f0.w);
            q0.z = pack_bf16_2(f1.x, f1.y); q0.w = pack_bf16_2(f1.z, f1.w);
            q1.x = pack_bf16_2(f2.x, f2.y); q1.y = pack_bf16_2(f2.z, f2.w);
            q1.z = pack_bf16_2(f3.x, f3.y); q1.w = pack_bf16_2(f3.z, f3.w);
            *(uint4*)&As[ldr * 40 + ldc]     = q0;
            *(uint4*)&As[ldr * 40 + ldc + 8] = q1;
        }
        {
            const float4* gp = (const float4*)(B + (size_t)(blockN + ldr) * K + kk + ldc);
            float4 f0 = gp[0], f1 = gp[1], f2 = gp[2], f3 = gp[3];
            uint4 q0, q1;
            q0.x = pack_bf16_2(f0.x, f0.y); q0.y = pack_bf16_2(f0.z, f0.w);
            q0.z = pack_bf16_2(f1.x, f1.y); q0.w = pack_bf16_2(f1.z, f1.w);
            q1.x = pack_bf16_2(f2.x, f2.y); q1.y = pack_bf16_2(f2.z, f2.w);
            q1.z = pack_bf16_2(f3.x, f3.y); q1.w = pack_bf16_2(f3.z, f3.w);
            *(uint4*)&Bs[ldr * 40 + ldc]     = q0;
            *(uint4*)&Bs[ldr * 40 + ldc + 8] = q1;
        }
        __syncthreads();

        v16bf af[2], bfg[4];
        #pragma unroll
        for (int fm = 0; fm < 2; ++fm)
            af[fm] = lds_frag(As, waveM * 32 + fm * 16 + l16, half);
        #pragma unroll
        for (int fn = 0; fn < 4; ++fn)
            bfg[fn] = lds_frag(Bs, waveN * 64 + fn * 16 + l16, half);

        #pragma unroll
        for (int fm = 0; fm < 2; ++fm)
            #pragma unroll
            for (int fn = 0; fn < 4; ++fn)
                acc[fm][fn] = __builtin_amdgcn_wmma_f32_16x16x32_bf16(
                    false, af[fm], false, bfg[fn],
                    (short)0, acc[fm][fn], false, false);
    }

    #pragma unroll
    for (int fn = 0; fn < 4; ++fn) {
        const int col = blockN + waveN * 64 + fn * 16 + l16;
        const float bv = bias[col];
        #pragma unroll
        for (int fm = 0; fm < 2; ++fm) {
            const int rowb = blockM + waveM * 32 + fm * 16 + half * 8;
            #pragma unroll
            for (int r = 0; r < 8; ++r) {
                float v = acc[fm][fn][r] + bv;
                if (ENC) v = (v > 0.0f) ? v : 0.0f;
                C[(size_t)(rowb + r) * N + col] = v;
            }
        }
    }
}

// ---------------------------------------------------------------------------
// one-time fp32 -> bf16 operand conversion (feeds the TDM path)
// ---------------------------------------------------------------------------
__global__ __launch_bounds__(256) void sae_conv_xc(
    const float4* __restrict__ x4, const float4* __restrict__ bdec4,
    uint2* __restrict__ out, size_t n4)
{
    const size_t stride = (size_t)gridDim.x * 256;
    for (size_t i = (size_t)blockIdx.x * 256 + threadIdx.x; i < n4; i += stride) {
        float4 v = x4[i];
        float4 b = bdec4[i & (INPUT_DIM / 4 - 1)];
        uint2 o;
        o.x = pack_bf16_2(v.x - b.x, v.y - b.y);
        o.y = pack_bf16_2(v.z - b.z, v.w - b.w);
        out[i] = o;
    }
}

__global__ __launch_bounds__(256) void sae_conv_w(
    const float4* __restrict__ in, uint2* __restrict__ out, size_t n4)
{
    const size_t stride = (size_t)gridDim.x * 256;
    for (size_t i = (size_t)blockIdx.x * 256 + threadIdx.x; i < n4; i += stride) {
        float4 v = in[i];
        uint2 o;
        o.x = pack_bf16_2(v.x, v.y);
        o.y = pack_bf16_2(v.z, v.w);
        out[i] = o;
    }
}

// ---------------------------------------------------------------------------
// Exact batch top-k threshold: 3-level radix select on uint bits (vals >= 0)
// ---------------------------------------------------------------------------
__global__ __launch_bounds__(256) void sae_hist0(
    const float* __restrict__ acts, unsigned* __restrict__ hist, size_t n4)
{
    __shared__ unsigned h[2048];
    for (int i = threadIdx.x; i < 2048; i += 256) h[i] = 0;
    __syncthreads();
    const float4* a4 = (const float4*)acts;
    const size_t stride = (size_t)gridDim.x * 256;
    for (size_t i = (size_t)blockIdx.x * 256 + threadIdx.x; i < n4; i += stride) {
        float4 v = a4[i];
        atomicAdd(&h[__float_as_uint(v.x) >> 20], 1u);
        atomicAdd(&h[__float_as_uint(v.y) >> 20], 1u);
        atomicAdd(&h[__float_as_uint(v.z) >> 20], 1u);
        atomicAdd(&h[__float_as_uint(v.w) >> 20], 1u);
    }
    __syncthreads();
    for (int i = threadIdx.x; i < 2048; i += 256)
        if (h[i]) atomicAdd(&hist[i], h[i]);
}

__global__ __launch_bounds__(256) void sae_hist_refine(
    const float* __restrict__ acts, const unsigned* __restrict__ ctrl,
    unsigned* __restrict__ hist, int level, size_t n4)
{
    __shared__ unsigned h[1024];
    for (int i = threadIdx.x; i < 1024; i += 256) h[i] = 0;
    __syncthreads();
    const unsigned match = (level == 1) ? ctrl[0] : ((ctrl[0] << 10) | ctrl[2]);
    const int keyshift = (level == 1) ? 20 : 10;
    const int bshift   = (level == 1) ? 10 : 0;
    const float4* a4 = (const float4*)acts;
    const size_t stride = (size_t)gridDim.x * 256;
    for (size_t i = (size_t)blockIdx.x * 256 + threadIdx.x; i < n4; i += stride) {
        float4 v = a4[i];
        unsigned b;
        b = __float_as_uint(v.x); if ((b >> keyshift) == match) atomicAdd(&h[(b >> bshift) & 1023], 1u);
        b = __float_as_uint(v.y); if ((b >> keyshift) == match) atomicAdd(&h[(b >> bshift) & 1023], 1u);
        b = __float_as_uint(v.z); if ((b >> keyshift) == match) atomicAdd(&h[(b >> bshift) & 1023], 1u);
        b = __float_as_uint(v.w); if ((b >> keyshift) == match) atomicAdd(&h[(b >> bshift) & 1023], 1u);
    }
    __syncthreads();
    for (int i = threadIdx.x; i < 1024; i += 256)
        if (h[i]) atomicAdd(&hist[i], h[i]);
}

__global__ void sae_scan(const unsigned* __restrict__ hist,
                         unsigned* __restrict__ ctrl, int level)
{
    if (threadIdx.x != 0) return;
    const unsigned target = (level == 0) ? TOPK_TOTAL : ctrl[(level == 1) ? 1 : 3];
    const int nb = (level == 0) ? 2048 : 1024;
    unsigned cum = 0;
    for (int b = nb - 1; b >= 0; --b) {
        unsigned c = hist[b];
        if (cum + c >= target) {
            if (level == 0)      { ctrl[0] = (unsigned)b; ctrl[1] = target - cum; }
            else if (level == 1) { ctrl[2] = (unsigned)b; ctrl[3] = target - cum; }
            else                 { ctrl[4] = (ctrl[0] << 20) | (ctrl[2] << 10) | (unsigned)b; }
            return;
        }
        cum += c;
    }
}

__global__ __launch_bounds__(256) void sae_apply(
    float* __restrict__ acts, const unsigned* __restrict__ ctrl,
    unsigned* __restrict__ cnt, size_t n4)
{
    const unsigned thr = ctrl[4];
    unsigned c = 0;
    float4* a4 = (float4*)acts;
    const size_t stride = (size_t)gridDim.x * 256;
    for (size_t i = (size_t)blockIdx.x * 256 + threadIdx.x; i < n4; i += stride) {
        float4 v = a4[i];
        unsigned b;
        b = __float_as_uint(v.x); { bool k = b >= thr; v.x = k ? v.x : 0.0f; c += (k && b != 0u); }
        b = __float_as_uint(v.y); { bool k = b >= thr; v.y = k ? v.y : 0.0f; c += (k && b != 0u); }
        b = __float_as_uint(v.z); { bool k = b >= thr; v.z = k ? v.z : 0.0f; c += (k && b != 0u); }
        b = __float_as_uint(v.w); { bool k = b >= thr; v.w = k ? v.w : 0.0f; c += (k && b != 0u); }
        a4[i] = v;
    }
    __shared__ unsigned s[256];
    s[threadIdx.x] = c;
    __syncthreads();
    for (int off = 128; off > 0; off >>= 1) {
        if ((int)threadIdx.x < off) s[threadIdx.x] += s[threadIdx.x + off];
        __syncthreads();
    }
    if (threadIdx.x == 0 && s[0]) atomicAdd(cnt, s[0]);
}

// ---------------------------------------------------------------------------
// loss: fixed-order two-stage reduction (bitwise deterministic)
// ---------------------------------------------------------------------------
__global__ __launch_bounds__(256) void sae_loss_partial(
    const float* __restrict__ recon, const float* __restrict__ x,
    float* __restrict__ partials, size_t n4)
{
    float s = 0.0f;
    const float4* r4 = (const float4*)recon;
    const float4* x4 = (const float4*)x;
    const size_t stride = (size_t)gridDim.x * 256;
    for (size_t i = (size_t)blockIdx.x * 256 + threadIdx.x; i < n4; i += stride) {
        float4 r = r4[i], xv = x4[i];
        float dx = r.x - xv.x, dy = r.y - xv.y, dz = r.z - xv.z, dw = r.w - xv.w;
        s += dx * dx + dy * dy + dz * dz + dw * dw;
    }
    __shared__ float sh[256];
    sh[threadIdx.x] = s;
    __syncthreads();
    for (int off = 128; off > 0; off >>= 1) {
        if ((int)threadIdx.x < off) sh[threadIdx.x] += sh[threadIdx.x + off];
        __syncthreads();
    }
    if (threadIdx.x == 0) partials[blockIdx.x] = sh[0];
}

__global__ __launch_bounds__(256) void sae_finalize(
    const float* __restrict__ partials, int nparts,
    const unsigned* __restrict__ ctrl, float* __restrict__ scalars)
{
    float v = 0.0f;
    for (int i = threadIdx.x; i < nparts; i += 256) v += partials[i];
    __shared__ float sh[256];
    sh[threadIdx.x] = v;
    __syncthreads();
    for (int off = 128; off > 0; off >>= 1) {
        if ((int)threadIdx.x < off) sh[threadIdx.x] += sh[threadIdx.x + off];
        __syncthreads();
    }
    if (threadIdx.x == 0) {
        float loss = sh[0] / (float)((size_t)BATCH * INPUT_DIM);
        scalars[0] = loss;
        scalars[1] = loss;
        scalars[2] = 0.0f;
        scalars[3] = (float)ctrl[5] / (float)BATCH;
    }
}

__global__ void sae_zero_ws(unsigned* __restrict__ ws, int n) {
    int i = blockIdx.x * 256 + threadIdx.x;
    if (i < n) ws[i] = 0;
}

// ---------------------------------------------------------------------------
extern "C" void kernel_launch(void* const* d_in, const int* in_sizes, int n_in,
                              void* d_out, int out_size, void* d_ws, size_t ws_size,
                              hipStream_t stream) {
    const float* x     = (const float*)d_in[0];
    const float* W_enc = (const float*)d_in[1];
    const float* b_enc = (const float*)d_in[2];
    const float* W_dec = (const float*)d_in[3];
    const float* b_dec = (const float*)d_in[4];

    float* recon   = (float*)d_out;
    float* feats   = recon + (size_t)BATCH * INPUT_DIM;
    float* scalars = feats + (size_t)BATCH * DICT_SIZE;

    // select/reduce area: first 32 KB of ws
    unsigned* wsu      = (unsigned*)d_ws;
    unsigned* hist0    = wsu;
    unsigned* hist1    = wsu + 2048;
    unsigned* hist2    = wsu + 3072;
    unsigned* ctrl     = wsu + 4096;
    float*    partials = (float*)(wsu + 4104);

    // bf16 operand area (TDM path)
    const size_t XC_E   = (size_t)BATCH * INPUT_DIM;      //  4M elems
    const size_t WENC_E = (size_t)DICT_SIZE * INPUT_DIM;  // 16M elems
    const size_t WDEC_E = WENC_E;
    bf16_t* xc_bf   = (bf16_t*)((char*)d_ws + 32768);
    bf16_t* wenc_bf = xc_bf + XC_E;
    bf16_t* wdec_bf = wenc_bf + WENC_E;
    const size_t WS_NEED = 32768 + (XC_E + WENC_E + WDEC_E) * sizeof(bf16_t);

    const size_t n4  = (size_t)BATCH * DICT_SIZE / 4;
    const size_t nr4 = (size_t)BATCH * INPUT_DIM / 4;

    const bool use_tdm = SAE_HAVE_TDM && (ws_size >= WS_NEED);

    if (use_tdm) {
        // one-time bf16 operand conversions (b_dec subtract folded into xc)
        sae_conv_xc<<<512, 256, 0, stream>>>((const float4*)x, (const float4*)b_dec,
                                             (uint2*)xc_bf, XC_E / 4);
        sae_conv_w<<<2048, 256, 0, stream>>>((const float4*)W_enc, (uint2*)wenc_bf, WENC_E / 4);
        sae_conv_w<<<2048, 256, 0, stream>>>((const float4*)W_dec, (uint2*)wdec_bf, WDEC_E / 4);
        // encoder: A (xc) and B (W_enc) both DMA'd by the Tensor Data Mover
        sae_gemm_wmma_tdm<true, true><<<dim3(DICT_SIZE / 128, BATCH / 128), 256, 0, stream>>>(
            xc_bf, nullptr, wenc_bf, b_enc, feats, BATCH, DICT_SIZE, INPUT_DIM);
    } else {
        sae_gemm_wmma_sync<true><<<dim3(DICT_SIZE / 128, BATCH / 128), 256, 0, stream>>>(
            x, W_enc, b_dec, b_enc, feats, BATCH, DICT_SIZE, INPUT_DIM);
    }

    sae_zero_ws<<<(5128 + 255) / 256, 256, 0, stream>>>(wsu, 5128);

    sae_hist0<<<2048, 256, 0, stream>>>(feats, hist0, n4);
    sae_scan<<<1, 32, 0, stream>>>(hist0, ctrl, 0);
    sae_hist_refine<<<2048, 256, 0, stream>>>(feats, ctrl, hist1, 1, n4);
    sae_scan<<<1, 32, 0, stream>>>(hist1, ctrl, 1);
    sae_hist_refine<<<2048, 256, 0, stream>>>(feats, ctrl, hist2, 2, n4);
    sae_scan<<<1, 32, 0, stream>>>(hist2, ctrl, 2);

    sae_apply<<<2048, 256, 0, stream>>>(feats, ctrl, ctrl + 5, n4);

    if (use_tdm) {
        // decoder: B (W_dec) via TDM; A (features, just produced) converted on the fly
        sae_gemm_wmma_tdm<false, false><<<dim3(INPUT_DIM / 128, BATCH / 128), 256, 0, stream>>>(
            nullptr, feats, wdec_bf, b_dec, recon, BATCH, INPUT_DIM, DICT_SIZE);
    } else {
        sae_gemm_wmma_sync<false><<<dim3(INPUT_DIM / 128, BATCH / 128), 256, 0, stream>>>(
            feats, W_dec, nullptr, b_dec, recon, BATCH, INPUT_DIM, DICT_SIZE);
    }

    sae_loss_partial<<<1024, 256, 0, stream>>>(recon, x, partials, nr4);
    sae_finalize<<<1, 256, 0, stream>>>(partials, 1024, ctrl, scalars);
}
